// CaMoE_System_18425409699868
// MI455X (gfx1250) — compile-verified, matrix operand order checked by reading, not run on hardware
//
#include <hip/hip_runtime.h>
#include <cstdint>
#include <cstddef>

// ---------------- problem dims (match reference) ----------------
constexpr int kB = 2, kT = 2048, kC = 1024, kV = 32000, kL = 4, kE = 3;
constexpr int kH = 4 * kC;          // 4096
constexpr int kM = kB * kT;         // 4096 tokens

typedef __attribute__((ext_vector_type(16))) __bf16 v16bf;
typedef __attribute__((ext_vector_type(8)))  float  v8f;
typedef __attribute__((ext_vector_type(4)))  int    v4i;
typedef unsigned short u16;

// ---------------- helpers ----------------
__device__ __forceinline__ u16 f2bf(float f) {
  union { float f; unsigned u; } x; x.f = f;
  unsigned r = (x.u + 0x7FFFu + ((x.u >> 16) & 1u)) >> 16;   // RNE
  return (u16)r;
}

// async global -> LDS copy of 16 bytes per lane (CDNA5 GLOBAL_LOAD_ASYNC_TO_LDS_B128)
// builtin signature (verified via hipcc): (int4 AS1*, int4 AS3*, imm, imm)
typedef __attribute__((address_space(1))) v4i gv4i;
typedef __attribute__((address_space(3))) v4i lv4i;

__device__ __forceinline__ void async_cp16(const void* g, void* l) {
#if __has_builtin(__builtin_amdgcn_global_load_async_to_lds_b128)
  __builtin_amdgcn_global_load_async_to_lds_b128((gv4i*)g, (lv4i*)l, 0, 0);
#else
  *reinterpret_cast<uint4*>(l) = *reinterpret_cast<const uint4*>(g);
#endif
}

template <int N>
__device__ __forceinline__ void wait_async() {
#if __has_builtin(__builtin_amdgcn_s_wait_asynccnt)
  __builtin_amdgcn_s_wait_asynccnt(N);
#else
  asm volatile("s_wait_asynccnt %0" :: "i"(N) : "memory");
#endif
}

// ---------------- WMMA bf16 GEMM:  C[M,N](f32) = A[M,K](bf16) * Bt[N,K](bf16)^T ----------------
constexpr int BM = 128, BN = 64, BK = 32, LDSW = BK + 8;   // +8 halfwords pad -> 80B row stride

union FragU { v16bf v; uint4 q[2]; };

__launch_bounds__(256)
__global__ void k_gemm_bf16(const u16* __restrict__ A, const u16* __restrict__ Bt,
                            float* __restrict__ Cm, int Mdim, int Ndim, int Kdim) {
  __shared__ u16 sA[2][BM][LDSW];
  __shared__ u16 sB[2][BN][LDSW];

  const int tid  = threadIdx.x;
  const int lane = tid & 31;
  const int wave = tid >> 5;          // 8 waves
  const int wm   = wave & 3;          // 4 row groups of 32
  const int wn   = wave >> 2;         // 2 col groups of 32
  const int tileM = blockIdx.y * BM;
  const int tileN = blockIdx.x * BN;

  // per-thread 16B chunk coordinates for tile staging
  const int ar0 = tid >> 2;                 // rows 0..63
  const int ar1 = (tid + 256) >> 2;         // rows 64..127
  const int ac  = (tid & 3) * 8;            // k sub-chunk
  const int br  = tid >> 2;                 // B rows 0..63

  const u16* Ag = A  + (size_t)tileM * Kdim;
  const u16* Bg = Bt + (size_t)tileN * Kdim;

  auto issue_stage = [&](int kpos, int buf) {
    async_cp16(Ag + (size_t)ar0 * Kdim + kpos + ac, &sA[buf][ar0][ac]);
    async_cp16(Ag + (size_t)ar1 * Kdim + kpos + ac, &sA[buf][ar1][ac]);
    async_cp16(Bg + (size_t)br  * Kdim + kpos + ac, &sB[buf][br ][ac]);
    // speculative L2 warm for the following stage (dropped if OOB per ISA)
    __builtin_prefetch(Ag + (size_t)ar0 * Kdim + kpos + BK + ac, 0, 1);
  };

  const v8f vzero = {0.f,0.f,0.f,0.f,0.f,0.f,0.f,0.f};
  v8f acc[2][2];
#pragma unroll
  for (int i = 0; i < 2; ++i)
#pragma unroll
    for (int j = 0; j < 2; ++j) acc[i][j] = vzero;

  const int r16   = lane & 15;
  const int half8 = (lane >> 4) * 8;

  auto compute = [&](int buf) {
    FragU fa[2], fb[2];
#pragma unroll
    for (int i = 0; i < 2; ++i) {
      const u16* p = &sA[buf][wm * 32 + i * 16 + r16][half8];
      fa[i].q[0] = *(const uint4*)p;
      fa[i].q[1] = *(const uint4*)(p + 16);
    }
#pragma unroll
    for (int j = 0; j < 2; ++j) {
      const u16* p = &sB[buf][wn * 32 + j * 16 + r16][half8];
      fb[j].q[0] = *(const uint4*)p;
      fb[j].q[1] = *(const uint4*)(p + 16);
    }
#pragma unroll
    for (int i = 0; i < 2; ++i)
#pragma unroll
      for (int j = 0; j < 2; ++j)
        acc[i][j] = __builtin_amdgcn_wmma_f32_16x16x32_bf16(
            false, fa[i].v, false, fb[j].v, (short)0, acc[i][j], false, false);
  };

  const int ksteps = Kdim / BK;       // >= 2 for every GEMM in this model
  issue_stage(0, 0);
  issue_stage(BK, 1);

  // steady state: wait for stage s (leaving stage s+1 in flight), compute, refill
  for (int s = 0; s + 2 < ksteps; ++s) {
    wait_async<3>();
    __syncthreads();
    compute(s & 1);
    __syncthreads();
    issue_stage((s + 2) * BK, s & 1);
  }
  // drain: stage ksteps-2 (stage ksteps-1 still in flight)
  wait_async<3>();
  __syncthreads();
  compute(ksteps & 1);
  __syncthreads();
  // drain: final stage
  wait_async<0>();
  __syncthreads();
  compute((ksteps - 1) & 1);

  // epilogue: C/D layout — lane 0..15 -> N=lane, M=vgpr; lane 16..31 -> M=8+vgpr
  const int n      = lane & 15;
  const int rowOff = (lane >> 4) * 8;
#pragma unroll
  for (int i = 0; i < 2; ++i)
#pragma unroll
    for (int j = 0; j < 2; ++j) {
      float* out = Cm + (size_t)(tileM + wm * 32 + i * 16 + rowOff) * Ndim
                      + (tileN + wn * 32 + j * 16 + n);
#pragma unroll
      for (int r = 0; r < 8; ++r) out[(size_t)r * Ndim] = acc[i][j][r];
    }
}

// ---------------- support kernels ----------------
__global__ void k_embed(const int* __restrict__ idx, const float* __restrict__ emb,
                        float* __restrict__ x) {
  const int tok = blockIdx.x;
  const int id  = idx[tok];
  for (int c = threadIdx.x; c < kC; c += 256)
    x[(size_t)tok * kC + c] = emb[(size_t)id * kC + c];
}

__global__ void k_ln(const float* __restrict__ x, const float* __restrict__ g,
                     const float* __restrict__ b, float* __restrict__ y) {
  __shared__ float s1[256], s2[256];
  const int tok = blockIdx.x, tid = threadIdx.x;
  const float* xp = x + (size_t)tok * kC;
  float sum = 0.f, sq = 0.f;
  for (int c = tid; c < kC; c += 256) { float v = xp[c]; sum += v; sq += v * v; }
  s1[tid] = sum; s2[tid] = sq; __syncthreads();
  for (int o = 128; o > 0; o >>= 1) {
    if (tid < o) { s1[tid] += s1[tid + o]; s2[tid] += s2[tid + o]; }
    __syncthreads();
  }
  const float mean = s1[0] / kC;
  const float var  = s2[0] / kC - mean * mean;
  const float inv  = rsqrtf(var + 1e-5f);
  for (int c = tid; c < kC; c += 256)
    y[(size_t)tok * kC + c] = (xp[c] - mean) * inv * g[c] + b[c];
}

__global__ void k_mix(const float* __restrict__ hs, u16* __restrict__ mixb, int n) {
  const int i = blockIdx.x * 256 + threadIdx.x;
  if (i >= n) return;
  const int tok = i / kC;
  const int t   = tok % kT;
  const float prev = (t == 0) ? 0.f : hs[i - kC];
  mixb[i] = f2bf(0.5f * (hs[i] + prev));
}

__global__ void k_transpose_bf16(const float* __restrict__ W, u16* __restrict__ Wt,
                                 int K, int N) {
  const int i = blockIdx.x * 256 + threadIdx.x;
  if (i >= K * N) return;
  const int k = i / N, n = i % N;
  Wt[(size_t)n * K + k] = f2bf(W[i]);
}

__global__ void k_f32_to_bf16(const float* __restrict__ a, u16* __restrict__ o, int n) {
  const int i = blockIdx.x * 256 + threadIdx.x;
  if (i < n) o[i] = f2bf(a[i]);
}

__global__ void k_relu_bf16(const float* __restrict__ a, u16* __restrict__ o, int n) {
  const int i = blockIdx.x * 256 + threadIdx.x;
  if (i < n) o[i] = f2bf(fmaxf(a[i], 0.f));
}

__global__ void k_cumsum(const float* __restrict__ kb, const float* __restrict__ vb,
                         float* __restrict__ state) {
  const int i = blockIdx.x * 256 + threadIdx.x;
  if (i >= kB * kC) return;
  const int b = i / kC, c = i % kC;
  float s = 0.f;
  for (int t = 0; t < kT; ++t) {
    const size_t p = ((size_t)(b * kT + t)) * kC + c;
    s += kb[p] * vb[p];
    state[p] = s / (float)(t + 1);
  }
}

__global__ void k_rs(const float* __restrict__ rlin, const float* __restrict__ state,
                     u16* __restrict__ rsb, int n) {
  const int i = blockIdx.x * 256 + threadIdx.x;
  if (i >= n) return;
  const float r = 1.f / (1.f + __expf(-rlin[i]));
  rsb[i] = f2bf(r * state[i]);
}

__global__ void k_add(float* __restrict__ x, const float* __restrict__ y, int n) {
  const int i = blockIdx.x * 256 + threadIdx.x;
  if (i < n) x[i] += y[i];
}

__global__ void k_hz(const float* __restrict__ h, const float* __restrict__ state,
                     u16* __restrict__ hz, int n) {
  const int i = blockIdx.x * 256 + threadIdx.x;
  if (i >= n) return;
  const int tok = i / (2 * kC), c = i % (2 * kC);
  hz[i] = f2bf(c < kC ? h[(size_t)tok * kC + c] : state[(size_t)tok * kC + (c - kC)]);
}

__global__ void k_route(const float* __restrict__ h, const float* __restrict__ wconf,
                        const float* __restrict__ Waff, const float* __restrict__ shares,
                        int* __restrict__ winners, float* __restrict__ scaleb) {
  const int tok = blockIdx.x * 256 + threadIdx.x;
  if (tok >= kM) return;
  float cf[kE] = {0.f, 0.f, 0.f}, af[kE] = {0.f, 0.f, 0.f};
  const float* hp = h + (size_t)tok * kC;
  for (int c = 0; c < kC; ++c) {
    const float hv = hp[c];
#pragma unroll
    for (int e = 0; e < kE; ++e) {
      cf[e] += hv * wconf[e * kC + c];
      af[e] += hv * Waff[c * kE + e];
    }
  }
  int w = 0; float best = -1e30f, wc = 0.f;
#pragma unroll
  for (int e = 0; e < kE; ++e) {
    const float conf = 1.f / (1.f + __expf(-cf[e]));
    const float bid  = conf * shares[e] + af[e];
    if (bid > best) { best = bid; w = e; wc = conf; }
  }
  winners[tok] = w;
  scaleb[tok]  = wc / (wc + 1e-6f);
}

__global__ void k_select(float* __restrict__ x, const float* __restrict__ f0,
                         const float* __restrict__ f1, const float* __restrict__ lin,
                         const int* __restrict__ winners, const float* __restrict__ scaleb,
                         int n) {
  const int i = blockIdx.x * 256 + threadIdx.x;
  if (i >= n) return;
  const int tok = i / kC;
  const int e = winners[tok];
  const float v = (e == 0) ? f0[i] : ((e == 1) ? f1[i] : lin[i]);
  x[i] += v * scaleb[tok];
}

// ---------------- host side ----------------
static inline dim3 ew(int n) { return dim3((n + 255) / 256); }

#define GEMM(Ab, Btb, Cb, Mm, Nn, Kk) \
  k_gemm_bf16<<<dim3((Nn) / BN, (Mm) / BM), 256, 0, stream>>>((Ab), (Btb), (Cb), (Mm), (Nn), (Kk))

extern "C" void kernel_launch(void* const* d_in, const int* in_sizes, int n_in,
                              void* d_out, int out_size, void* d_ws, size_t ws_size,
                              hipStream_t stream) {
  (void)in_sizes; (void)n_in; (void)out_size; (void)ws_size;
  const int*   idx    = (const int*)  d_in[0];
  const float* shares = (const float*)d_in[1];
  const float* emb    = (const float*)d_in[2];
  const float* ln1_g  = (const float*)d_in[3];
  const float* ln1_b  = (const float*)d_in[4];
  const float* ln2_g  = (const float*)d_in[5];
  const float* ln2_b  = (const float*)d_in[6];
  const float* Wr     = (const float*)d_in[7];
  const float* Wk     = (const float*)d_in[8];
  const float* Wv     = (const float*)d_in[9];
  const float* Wo     = (const float*)d_in[10];
  const float* W1     = (const float*)d_in[11];
  const float* W2     = (const float*)d_in[12];
  const float* wconf  = (const float*)d_in[13];
  const float* Wl1    = (const float*)d_in[14];
  const float* Wl2    = (const float*)d_in[15];
  const float* lnf_g  = (const float*)d_in[18];
  const float* lnf_b  = (const float*)d_in[19];
  const float* headW  = (const float*)d_in[20];
  const float* Waff   = (const float*)d_in[17];   // d_in[16] = W_diff (unused in fwd)
  float* logits = (float*)d_out;

  // bump allocator over workspace
  char* base = (char*)d_ws;
  size_t off = 0;
  auto alloc = [&](size_t bytes) -> void* {
    void* p = base + off;
    off += (bytes + 255) & ~(size_t)255;
    return p;
  };
  const size_t MC = (size_t)kM * kC, MH = (size_t)kM * kH;
  float* xbuf  = (float*)alloc(MC * 4);
  float* hs    = (float*)alloc(MC * 4);
  float* hbuf  = (float*)alloc(MC * 4);
  float* state = (float*)alloc(MC * 4);
  float* rlin  = (float*)alloc(MC * 4);
  float* kbuf  = (float*)alloc(MC * 4);
  float* vbuf  = (float*)alloc(MC * 4);
  float* ffn0  = (float*)alloc(MC * 4);
  float* ffn1  = (float*)alloc(MC * 4);
  float* linb  = (float*)alloc(MC * 4);
  float* tmpC  = (float*)alloc(MC * 4);
  float* t1    = (float*)alloc(MH * 4);
  u16* mixb = (u16*)alloc(MC * 2);
  u16* rsb  = (u16*)alloc(MC * 2);
  u16* hb   = (u16*)alloc(MC * 2);
  u16* xfb  = (u16*)alloc(MC * 2);
  u16* hzb  = (u16*)alloc(2 * MC * 2);
  u16* t1b  = (u16*)alloc(MH * 2);
  u16* wtb  = (u16*)alloc((size_t)kV * kC * 2);   // shared transposed-weight buffer (max = head)
  int*   winners = (int*)  alloc((size_t)kM * 4);
  float* scaleb  = (float*)alloc((size_t)kM * 4);

  k_embed<<<kM, 256, 0, stream>>>(idx, emb, xbuf);

  for (int l = 0; l < kL; ++l) {
    // --- time-mix ---
    k_ln<<<kM, 256, 0, stream>>>(xbuf, ln1_g + l * kC, ln1_b + l * kC, hs);
    k_mix<<<ew(MC), 256, 0, stream>>>(hs, mixb, (int)MC);

    k_transpose_bf16<<<ew(kC * kC), 256, 0, stream>>>(Wr + (size_t)l * kC * kC, wtb, kC, kC);
    GEMM(mixb, wtb, rlin, kM, kC, kC);
    k_transpose_bf16<<<ew(kC * kC), 256, 0, stream>>>(Wk + (size_t)l * kC * kC, wtb, kC, kC);
    GEMM(mixb, wtb, kbuf, kM, kC, kC);
    k_transpose_bf16<<<ew(kC * kC), 256, 0, stream>>>(Wv + (size_t)l * kC * kC, wtb, kC, kC);
    GEMM(mixb, wtb, vbuf, kM, kC, kC);

    k_cumsum<<<ew(kB * kC), 256, 0, stream>>>(kbuf, vbuf, state);
    k_rs<<<ew(MC), 256, 0, stream>>>(rlin, state, rsb, (int)MC);

    k_transpose_bf16<<<ew(kC * kC), 256, 0, stream>>>(Wo + (size_t)l * kC * kC, wtb, kC, kC);
    GEMM(rsb, wtb, tmpC, kM, kC, kC);
    k_add<<<ew(MC), 256, 0, stream>>>(xbuf, tmpC, (int)MC);

    // --- routing ---
    k_ln<<<kM, 256, 0, stream>>>(xbuf, ln2_g + l * kC, ln2_b + l * kC, hbuf);
    k_f32_to_bf16<<<ew(MC), 256, 0, stream>>>(hbuf, hb, (int)MC);
    k_route<<<ew(kM), 256, 0, stream>>>(hbuf, wconf + (size_t)l * kE * kC,
                                        Waff + (size_t)l * kC * kE, shares + l * kE,
                                        winners, scaleb);

    // --- FFN experts ---
    for (int ex = 0; ex < 2; ++ex) {
      k_transpose_bf16<<<ew(kC * kH), 256, 0, stream>>>(
          W1 + ((size_t)(l * 2 + ex)) * kC * kH, wtb, kC, kH);
      GEMM(hb, wtb, t1, kM, kH, kC);
      k_relu_bf16<<<ew(MH), 256, 0, stream>>>(t1, t1b, (int)MH);
      k_transpose_bf16<<<ew(kH * kC), 256, 0, stream>>>(
          W2 + ((size_t)(l * 2 + ex)) * kH * kC, wtb, kH, kC);
      GEMM(t1b, wtb, (ex == 0 ? ffn0 : ffn1), kM, kC, kH);
    }

    // --- linear-transformer expert on (h, state) ---
    k_hz<<<ew(2 * MC), 256, 0, stream>>>(hbuf, state, hzb, (int)(2 * MC));
    k_transpose_bf16<<<ew(2 * kC * kH), 256, 0, stream>>>(
        Wl1 + (size_t)l * 2 * kC * kH, wtb, 2 * kC, kH);
    GEMM(hzb, wtb, t1, kM, kH, 2 * kC);
    k_relu_bf16<<<ew(MH), 256, 0, stream>>>(t1, t1b, (int)MH);
    k_transpose_bf16<<<ew(kH * kC), 256, 0, stream>>>(
        Wl2 + (size_t)l * kH * kC, wtb, kH, kC);
    GEMM(t1b, wtb, linb, kM, kC, kH);

    k_select<<<ew(MC), 256, 0, stream>>>(xbuf, ffn0, ffn1, linb, winners, scaleb, (int)MC);
  }

  // --- head ---
  k_ln<<<kM, 256, 0, stream>>>(xbuf, lnf_g, lnf_b, hs);
  k_f32_to_bf16<<<ew(MC), 256, 0, stream>>>(hs, xfb, (int)MC);
  k_transpose_bf16<<<ew(kC * kV), 256, 0, stream>>>(headW, wtb, kC, kV);
  GEMM(xfb, wtb, logits, kM, kV, kC);
}
